// Causal_Attention_27659589386607
// MI455X (gfx1250) — matrix operand order repaired and checked by reading, not compiled, not run on hardware
//
#include <hip/hip_runtime.h>

// ---------------------------------------------------------------------------
// Causal attention for B=4, T=4096, C=1024, H=64 on gfx1250 (MI455X).
// Kernel 1: fused QKV projection, fp32 in -> bf16 out, via v_wmma_f32_16x16x32_bf16
// Kernel 2: flash attention (online softmax, fp32 accum), all matmuls on WMMA;
//           K tiles staged LDS via the Tensor Data Mover (tensor_load_to_lds).
// ---------------------------------------------------------------------------

#define BATCH 4
#define T_SEQ 4096
#define CDIM  1024
#define HDIM  64

typedef __attribute__((ext_vector_type(16))) __bf16        v16bf;
typedef __attribute__((ext_vector_type(8)))  float         v8f;
typedef __attribute__((ext_vector_type(4)))  unsigned int  u32x4;
typedef __attribute__((ext_vector_type(8)))  unsigned int  u32x8;
typedef __attribute__((ext_vector_type(4)))  float         f32x4;

union FragU {            // bit-cast between 16xbf16 fragment and two 128-bit loads
    v16bf v;
    u32x4 q[2];
    unsigned short s[16];
};

__device__ __forceinline__ unsigned short f2bf(float f) {
    union { __bf16 h; unsigned short s; } u;
    u.h = (__bf16)f;
    return u.s;
}

// ---------------------------------------------------------------------------
// Kernel 1: Q/K/V = x @ W^T   (x: [B*T, C] f32, W: [H, C] f32, out: [B*T, H] bf16)
// Block = 128 threads (4 waves). Block owns 16 rows; wave w owns cols [16w,16w+16)
// of each of Q, K, V. Q is pre-scaled by 1/sqrt(H) so the flash kernel skips it.
// ---------------------------------------------------------------------------
__global__ __launch_bounds__(128) void qkv_proj_kernel(
    const float* __restrict__ x,
    const float* __restrict__ Wq,
    const float* __restrict__ Wk,
    const float* __restrict__ Wv,
    unsigned short* __restrict__ Qb,
    unsigned short* __restrict__ Kb,
    unsigned short* __restrict__ Vb)
{
    __shared__ float xs[16 * 32];              // 2 KB staging tile of x

    const int tid  = threadIdx.x;
    const int lane = tid & 31;
    const int wave = tid >> 5;                 // 0..3
    const int l15  = lane & 15;
    const int hi   = lane >> 4;                // 0 or 1

    const int rowBase = blockIdx.x * 16;       // rows of [B*T]
    const int nBase   = wave * 16;             // output column slice

    v8f accQ = {}; v8f accK = {}; v8f accV = {};

    for (int kc = 0; kc < CDIM; kc += 32) {
        // ---- stage x[rowBase:+16][kc:+32] into LDS (coalesced f32x4) ----
        {
            const int r = tid >> 3;            // 0..15
            const int c = (tid & 7) * 4;       // 0..28
            f32x4 vv = *(const f32x4*)(x + (size_t)(rowBase + r) * CDIM + kc + c);
            *(f32x4*)&xs[r * 32 + c] = vv;
            if (kc + 32 < CDIM)                // prefetch next tile (global_prefetch_b8)
                __builtin_prefetch(x + (size_t)(rowBase + r) * CDIM + kc + 32 + c, 0, 1);
        }
        __syncthreads();

        // ---- A fragment (16x32, bf16): lane = row, hi-half holds K+16 ----
        v16bf a;
        {
            const int r  = l15;
            const int k0 = hi * 16;
            #pragma unroll
            for (int j = 0; j < 8; ++j) {
                a[j]     = (__bf16)xs[r * 32 + k0 + j];
                a[8 + j] = (__bf16)xs[r * 32 + k0 + 8 + j];
            }
        }

        // ---- B fragments (32x16): lane = out column, hi-half holds K+16 ----
        const size_t wOff = (size_t)(nBase + l15) * CDIM + kc + hi * 16;
        v16bf bq, bk, bv;
        #pragma unroll
        for (int q4 = 0; q4 < 4; ++q4) {
            f32x4 wq4 = *(const f32x4*)(Wq + wOff + q4 * 4);
            f32x4 wk4 = *(const f32x4*)(Wk + wOff + q4 * 4);
            f32x4 wv4 = *(const f32x4*)(Wv + wOff + q4 * 4);
            #pragma unroll
            for (int j = 0; j < 4; ++j) {
                bq[q4 * 4 + j] = (__bf16)wq4[j];
                bk[q4 * 4 + j] = (__bf16)wk4[j];
                bv[q4 * 4 + j] = (__bf16)wv4[j];
            }
        }

        accQ = __builtin_amdgcn_wmma_f32_16x16x32_bf16(false, a, false, bq, (short)0, accQ, false, false);
        accK = __builtin_amdgcn_wmma_f32_16x16x32_bf16(false, a, false, bk, (short)0, accK, false, false);
        accV = __builtin_amdgcn_wmma_f32_16x16x32_bf16(false, a, false, bv, (short)0, accV, false, false);

        __syncthreads();
    }

    // ---- store bf16: C layout -> vgpr r is row r + 8*hi, lane&15 is col ----
    #pragma unroll
    for (int r = 0; r < 8; ++r) {
        const size_t o = (size_t)(rowBase + r + 8 * hi) * HDIM + nBase + l15;
        Qb[o] = f2bf(accQ[r] * 0.125f);        // fold 1/sqrt(64) into Q
        Kb[o] = f2bf(accK[r]);
        Vb[o] = f2bf(accV[r]);
    }
}

// ---------------------------------------------------------------------------
// Kernel 2: causal flash attention. Block = 128 threads = 64 queries (16/wave).
// Key blocks of 32; K tile DMA'd into LDS by the Tensor Data Mover; S via 4
// WMMAs, online softmax in fp32, P re-shaped through per-wave LDS, P*V via 4
// WMMAs into 4 fp32 accumulators (h = 64).
// ---------------------------------------------------------------------------
__global__ __launch_bounds__(128) void flash_attn_kernel(
    const unsigned short* __restrict__ Qb,
    const unsigned short* __restrict__ Kb,
    const unsigned short* __restrict__ Vb,
    float* __restrict__ out)
{
    __shared__ unsigned short Ks[32 * 64];     // [key][h]    4 KB (TDM destination)
    __shared__ unsigned short VTs[64 * 32];    // [h][key]    4 KB (transposed V)
    __shared__ float          Ps[4][16 * 32];  // per-wave P  8 KB

    const int tid  = threadIdx.x;
    const int lane = tid & 31;
    const int wave = tid >> 5;
    const int l15  = lane & 15;
    const int hi   = lane >> 4;

    const int tilesPerBatch = T_SEQ / 64;
    const int b       = blockIdx.x / tilesPerBatch;
    const int qt      = blockIdx.x % tilesPerBatch;
    const int qbaseWG = qt * 64;
    const int qbase   = qbaseWG + wave * 16;

    const size_t batchOff = (size_t)b * T_SEQ * HDIM;

    // ---- Q fragments for this wave: two 16x32 bf16 A fragments (h 0..31, 32..63)
    v16bf qa[2];
    {
        const unsigned short* qp = Qb + batchOff + (size_t)(qbase + l15) * HDIM;
        #pragma unroll
        for (int Hh = 0; Hh < 2; ++Hh) {
            FragU f;
            const unsigned short* p = qp + Hh * 32 + hi * 16;
            f.q[0] = *(const u32x4*)(p);
            f.q[1] = *(const u32x4*)(p + 8);
            qa[Hh] = f.v;
        }
    }

    // Constant part of the TDM descriptor (wave-uniform SGPR data).
    const unsigned int ldsKAddr = (unsigned int)(size_t)&Ks[0];   // LDS byte offset

    float mrow[8], lrow[8];
    #pragma unroll
    for (int r = 0; r < 8; ++r) { mrow[r] = -1e30f; lrow[r] = 0.0f; }
    v8f o[4] = {v8f{}, v8f{}, v8f{}, v8f{}};

    const int nblocks = (qbaseWG + 64) / 32;   // cover the block's diagonal

    for (int kb = 0; kb < nblocks; ++kb) {
        const int kbase = kb * 32;

        // ---- TDM: DMA K[kbase:+32][0:64] (bf16) into Ks, issued by wave 0 ----
        if (wave == 0) {
            const unsigned long long gaddr =
                (unsigned long long)(Kb + batchOff + (size_t)kbase * HDIM);
            u32x4 g0;
            g0[0] = 1u;                                        // count=1 (valid D#)
            g0[1] = ldsKAddr;                                  // lds_addr
            g0[2] = (unsigned int)(gaddr & 0xFFFFFFFFu);       // global_addr[31:0]
            g0[3] = (unsigned int)((gaddr >> 32) & 0x01FFFFFFu)
                  | (2u << 30);                                // global_addr[56:32] | type=2
            u32x8 g1;
            g1[0] = 1u << 16;                                  // data_size=1 (2 bytes)
            g1[1] = (unsigned int)HDIM << 16;                  // tensor_dim0 = 64
            g1[2] = (unsigned int)T_SEQ << 16;                 // tensor_dim1 = 4096
            g1[3] = (unsigned int)HDIM << 16;                  // tile_dim0 = 64
            g1[4] = 32u;                                       // tile_dim1 = 32
            g1[5] = (unsigned int)HDIM;                        // tensor_dim0_stride = 64
            g1[6] = 0u;
            g1[7] = 0u;
            asm volatile("tensor_load_to_lds %0, %1"
                         :: "s"(g0), "s"(g1)
                         : "memory");
            __builtin_amdgcn_s_wait_tensorcnt(0);
        }

        // ---- cooperative staging of transposed V ----
        {
            const int r = tid >> 2;            // key row 0..31
            const int c = (tid & 3) * 16;      // h col 0,16,32,48
            const unsigned short* vp = Vb + batchOff + (size_t)(kbase + r) * HDIM + c;
            u32x4 v0 = *(const u32x4*)(vp);
            u32x4 v1 = *(const u32x4*)(vp + 8);
            if (kb + 1 < nblocks)              // prefetch next V tile
                __builtin_prefetch(Vb + batchOff + (size_t)(kbase + 32 + r) * HDIM + c, 0, 1);
            const unsigned short* ve0 = (const unsigned short*)&v0;
            const unsigned short* ve1 = (const unsigned short*)&v1;
            #pragma unroll
            for (int j = 0; j < 8; ++j) {
                VTs[(c + j) * 32 + r]     = ve0[j];
                VTs[(c + 8 + j) * 32 + r] = ve1[j];
            }
        }
        __syncthreads();

        if (kbase <= qbase + 15) {             // wave-uniform: EXEC stays all-ones
            // ---- S = Q K^T : two key halves x two h chunks = 4 WMMAs ----
            v8f s0 = {}, s1 = {};
            #pragma unroll
            for (int Hh = 0; Hh < 2; ++Hh) {
                FragU bk0, bk1;
                const int base0 = (l15)      * 64 + Hh * 32 + hi * 16;
                const int base1 = (16 + l15) * 64 + Hh * 32 + hi * 16;
                bk0.q[0] = *(const u32x4*)&Ks[base0];
                bk0.q[1] = *(const u32x4*)&Ks[base0 + 8];
                bk1.q[0] = *(const u32x4*)&Ks[base1];
                bk1.q[1] = *(const u32x4*)&Ks[base1 + 8];
                s0 = __builtin_amdgcn_wmma_f32_16x16x32_bf16(false, qa[Hh], false, bk0.v, (short)0, s0, false, false);
                s1 = __builtin_amdgcn_wmma_f32_16x16x32_bf16(false, qa[Hh], false, bk1.v, (short)0, s1, false, false);
            }

            // ---- online softmax (rows live in 16-lane halves; Q pre-scaled) ----
            float* Pw = &Ps[wave][0];
            #pragma unroll
            for (int r = 0; r < 8; ++r) {
                const int qrow = qbase + r + 8 * hi;
                const int kc0  = kbase + l15;
                const int kc1  = kbase + 16 + l15;
                float v0 = (kc0 <= qrow) ? s0[r] : -1e30f;
                float v1 = (kc1 <= qrow) ? s1[r] : -1e30f;

                float vmax = fmaxf(v0, v1);
                vmax = fmaxf(vmax, __shfl_xor(vmax, 1, 32));
                vmax = fmaxf(vmax, __shfl_xor(vmax, 2, 32));
                vmax = fmaxf(vmax, __shfl_xor(vmax, 4, 32));
                vmax = fmaxf(vmax, __shfl_xor(vmax, 8, 32));
                const float mnew = fmaxf(mrow[r], vmax);

                const float p0 = __expf(v0 - mnew);
                const float p1 = __expf(v1 - mnew);
                float ps = p0 + p1;
                ps += __shfl_xor(ps, 1, 32);
                ps += __shfl_xor(ps, 2, 32);
                ps += __shfl_xor(ps, 4, 32);
                ps += __shfl_xor(ps, 8, 32);

                const float corr = __expf(mrow[r] - mnew);
                lrow[r] = lrow[r] * corr + ps;
                mrow[r] = mnew;
                o[0][r] *= corr; o[1][r] *= corr; o[2][r] *= corr; o[3][r] *= corr;

                const int m = r + 8 * hi;
                Pw[m * 32 + l15]      = p0;
                Pw[m * 32 + 16 + l15] = p1;
            }

            // ---- re-shape P (C layout -> A layout) via per-wave LDS ----
            FragU pf;
            {
                const int k0 = hi * 16;
                #pragma unroll
                for (int j = 0; j < 8; ++j) {
                    pf.v[j]     = (__bf16)Pw[l15 * 32 + k0 + j];
                    pf.v[8 + j] = (__bf16)Pw[l15 * 32 + k0 + 8 + j];
                }
            }

            // ---- O += P * V : 4 WMMAs over h tiles ----
            #pragma unroll
            for (int t = 0; t < 4; ++t) {
                FragU bv;
                const int vb = (t * 16 + l15) * 32 + hi * 16;
                bv.q[0] = *(const u32x4*)&VTs[vb];
                bv.q[1] = *(const u32x4*)&VTs[vb + 8];
                o[t] = __builtin_amdgcn_wmma_f32_16x16x32_bf16(false, pf.v, false, bv.v, (short)0, o[t], false, false);
            }
        }
        __syncthreads();
    }

    // ---- normalize and store fp32 output ----
    float* op = out + batchOff;
    #pragma unroll
    for (int r = 0; r < 8; ++r) {
        const float inv  = 1.0f / lrow[r];
        const int   qrow = qbase + r + 8 * hi;
        #pragma unroll
        for (int t = 0; t < 4; ++t) {
            op[(size_t)qrow * HDIM + t * 16 + l15] = o[t][r] * inv;
        }
    }
}

// ---------------------------------------------------------------------------
extern "C" void kernel_launch(void* const* d_in, const int* in_sizes, int n_in,
                              void* d_out, int out_size, void* d_ws, size_t ws_size,
                              hipStream_t stream) {
    (void)in_sizes; (void)n_in; (void)out_size; (void)ws_size;

    const float* x  = (const float*)d_in[0];
    const float* Wq = (const float*)d_in[1];
    const float* Wk = (const float*)d_in[2];
    const float* Wv = (const float*)d_in[3];
    float* out = (float*)d_out;

    const size_t nBT = (size_t)BATCH * T_SEQ * HDIM;   // 1,048,576 bf16 elems each
    unsigned short* Qb = (unsigned short*)d_ws;
    unsigned short* Kb = Qb + nBT;
    unsigned short* Vb = Kb + nBT;

    qkv_proj_kernel<<<(BATCH * T_SEQ) / 16, 128, 0, stream>>>(x, Wq, Wk, Wv, Qb, Kb, Vb);
    flash_attn_kernel<<<BATCH * (T_SEQ / 64), 128, 0, stream>>>(Qb, Kb, Vb, out);
}
